// OutputRBFUnits_12446815224431
// MI455X (gfx1250) — matrix operand order repaired and checked by reading, not compiled, not run on hardware
//
#include <hip/hip_runtime.h>

#define BATCH       32768
#define IN_DIM      64
#define NUM_CLASSES 4096

typedef __attribute__((ext_vector_type(16))) _Float16 v16h;
typedef __attribute__((ext_vector_type(8)))  _Float16 v8h;
typedef __attribute__((ext_vector_type(8)))  float    v8f;
typedef __attribute__((ext_vector_type(4)))  int      v4i;
typedef __attribute__((address_space(3)))    v4i      lds_v4i;

// ---------------------------------------------------------------------------
// CDNA5 async-to-LDS path (ASYNCcnt). Probe-learned signature:
//   __builtin_amdgcn_global_load_async_to_lds_b128(v4i* gsrc,
//       __attribute__((address_space(3))) v4i* lds, imm offset, imm cpol)
// (param0 from round-3 diagnostic, param1 from round-4 diagnostic).
// ---------------------------------------------------------------------------
#if defined(__gfx1250__) && __has_builtin(__builtin_amdgcn_global_load_async_to_lds_b128)
#define HAVE_ASYNC_LDS 1
#else
#define HAVE_ASYNC_LDS 0
#endif

#if defined(__gfx1250__) && __has_builtin(__builtin_amdgcn_s_wait_asynccnt)
#define WAIT_ASYNC(n) __builtin_amdgcn_s_wait_asynccnt(n)
#else
#define WAIT_ASYNC(n) asm volatile("s_wait_asynccnt %0" ::"i"(n) : "memory")
#endif

// ---------------------------------------------------------------------------
// Workspace layout (bytes). Total ~8.8 MB.
// ---------------------------------------------------------------------------
#define WS_XSQ   0                                      // BATCH f32
#define WS_WSQ   (WS_XSQ + BATCH * 4)                   // NUM_CLASSES f32
#define WS_XHI   (WS_WSQ + NUM_CLASSES * 4)             // BATCH*IN_DIM f16
#define WS_XLO   (WS_XHI + BATCH * IN_DIM * 2)          // BATCH*IN_DIM f16
#define WS_WH    (WS_XLO + BATCH * IN_DIM * 2)          // NUM_CLASSES*IN_DIM f16

// ---------------------------------------------------------------------------
// Prep 1: x -> (x_hi + x_lo) f16 split (~22-bit effective mantissa) + ||x||^2.
// ---------------------------------------------------------------------------
__global__ void rbf_prep_x(const float* __restrict__ x, float* __restrict__ xsq,
                           _Float16* __restrict__ xhi, _Float16* __restrict__ xlo) {
    const int row = blockIdx.x * blockDim.x + threadIdx.x;
    const float4* xr = (const float4*)(x + (size_t)row * IN_DIM);
    v8h* hip = (v8h*)(xhi + (size_t)row * IN_DIM);
    v8h* lop = (v8h*)(xlo + (size_t)row * IN_DIM);
    float s = 0.f;
#pragma unroll
    for (int i = 0; i < IN_DIM / 8; ++i) {
        float4 a = xr[2 * i], b = xr[2 * i + 1];
        float f[8] = {a.x, a.y, a.z, a.w, b.x, b.y, b.z, b.w};
        v8h h, l;
#pragma unroll
        for (int j = 0; j < 8; ++j) {
            s += f[j] * f[j];
            _Float16 hh = (_Float16)f[j];
            h[j] = hh;
            l[j] = (_Float16)(f[j] - (float)hh);
        }
        hip[i] = h;
        lop[i] = l;
    }
    xsq[row] = s;
}

// ---------------------------------------------------------------------------
// Prep 2: w -> f16 (exact: entries are +/-1) + ||w||^2.
// ---------------------------------------------------------------------------
__global__ void rbf_prep_w(const float* __restrict__ w, float* __restrict__ wsq,
                           _Float16* __restrict__ wh) {
    const int row = blockIdx.x * blockDim.x + threadIdx.x;
    const float4* wr = (const float4*)(w + (size_t)row * IN_DIM);
    v8h* hp = (v8h*)(wh + (size_t)row * IN_DIM);
    float s = 0.f;
#pragma unroll
    for (int i = 0; i < IN_DIM / 8; ++i) {
        float4 a = wr[2 * i], b = wr[2 * i + 1];
        float f[8] = {a.x, a.y, a.z, a.w, b.x, b.y, b.z, b.w};
        v8h h;
#pragma unroll
        for (int j = 0; j < 8; ++j) {
            s += f[j] * f[j];
            h[j] = (_Float16)f[j];
        }
        hp[i] = h;
    }
    wsq[row] = s;
}

// ---------------------------------------------------------------------------
// GEMM: logits = 2*(x.w) - ||x||^2 - ||w||^2.
// Grid (4, 256); block 256 = 8 waves; workgroup sweeps 8 column blocks of 128.
// A tile (hi/lo f16) + xsq resident in registers. w block staged in LDS:
//   async path : double-buffered, global_load_async_to_lds_b128 (ASYNCcnt)
//   fallback   : single buffer via registers (round-2 scheme)
// ---------------------------------------------------------------------------
#define LDSB_HALVES (128 * 4 * 16)   // 16 KB per buffer

__global__ __launch_bounds__(256) void rbf_gemm_kernel(
        const _Float16* __restrict__ xhi, const _Float16* __restrict__ xlo,
        const _Float16* __restrict__ wh,
        const float* __restrict__ xsq, const float* __restrict__ wsq,
        float* __restrict__ out) {

#if HAVE_ASYNC_LDS
    __shared__ __align__(64) _Float16 ldsB[2 * LDSB_HALVES];   // 32 KB, 2 buffers
#else
    __shared__ __align__(64) _Float16 ldsB[LDSB_HALVES];       // 16 KB
#endif

    const int tid  = threadIdx.x;
    const int lane = tid & 31;
    const int wave = tid >> 5;
    const int g    = lane >> 4;      // half-wave group (0/1)
    const int nl   = lane & 15;

    const int rowBase  = blockIdx.y * 128;
    const int colChunk = blockIdx.x * 1024;

    // ---- resident A tile: 16 rows x 64 K element s holds k = 32c + 16g + s (contiguous order, same as B).
    const size_t arow = (size_t)(rowBase + wave * 16 + nl) * IN_DIM;
    const int kb = g * 16;
    v16h a_hi[2], a_lo[2];
#pragma unroll
    for (int c = 0; c < 2; ++c) {
        v8h h0 = *(const v8h*)(xhi + arow + 32 * c + kb);
        v8h h1 = *(const v8h*)(xhi + arow + 32 * c + kb + 8);
        v8h l0 = *(const v8h*)(xlo + arow + 32 * c + kb);
        v8h l1 = *(const v8h*)(xlo + arow + 32 * c + kb + 8);
        a_hi[c] = __builtin_shufflevector(h0, h1, 0,1,2,3,4,5,6,7,8,9,10,11,12,13,14,15);
        a_lo[c] = __builtin_shufflevector(l0, l1, 0,1,2,3,4,5,6,7,8,9,10,11,12,13,14,15);
    }

    float xs[8];
#pragma unroll
    for (int v = 0; v < 8; ++v)
        xs[v] = xsq[rowBase + wave * 16 + 8 * g + v];   // half-wave broadcast load

    // Each thread owns one contiguous 64 B chunk of the w block per step:
    // records 2*tid, 2*tid+1  <->  wh[(colBase + tid/2)*64 + (tid&1)*32 .. +31]
    const size_t woff = (size_t)(tid >> 1) * IN_DIM + (size_t)(tid & 1) * 32;

    // ---- 16x16-tile compute + epilogue for one 128-column block ----
    auto compute_block = [&](const v16h* Bv, int cb) {
        const v8f zero = {};
        v8f acc[8];
#pragma unroll
        for (int t = 0; t < 8; ++t) acc[t] = zero;

#pragma unroll
        for (int t = 0; t < 8; ++t) {
            const int n_loc = 16 * t + nl;
            v16h b0 = Bv[n_loc * 4 + 0 + g];   // K chunk 0 (quarters 0/1)
            v16h b1 = Bv[n_loc * 4 + 2 + g];   // K chunk 1 (quarters 2/3)
            acc[t] = __builtin_amdgcn_wmma_f32_16x16x32_f16(
                false, a_hi[0], false, b0, (short)0, acc[t], false, false);
            acc[t] = __builtin_amdgcn_wmma_f32_16x16x32_f16(
                false, a_lo[0], false, b0, (short)0, acc[t], false, false);
            acc[t] = __builtin_amdgcn_wmma_f32_16x16x32_f16(
                false, a_hi[1], false, b1, (short)0, acc[t], false, false);
            acc[t] = __builtin_amdgcn_wmma_f32_16x16x32_f16(
                false, a_lo[1], false, b1, (short)0, acc[t], false, false);
        }

        // epilogue: D layout lane L, VGPR v -> (M = v + 8g, N = nl)
        const int colBase = colChunk + cb * 128;
#pragma unroll
        for (int t = 0; t < 8; ++t) {
            const int col = colBase + 16 * t + nl;
            const float wq = wsq[col];
            float* orow = out + (size_t)(rowBase + wave * 16 + 8 * g) * NUM_CLASSES + col;
#pragma unroll
            for (int v = 0; v < 8; ++v)
                orow[(size_t)v * NUM_CLASSES] = 2.f * acc[t][v] - xs[v] - wq;
        }
    };

#if HAVE_ASYNC_LDS
    // ---- async path: ISA adds INST_OFFSET to BOTH global and LDS addresses,
    //      so one base pair + imm offsets copies a contiguous 64 B chunk.
#define ASYNC_FILL(buf, cbi)                                                          \
    do {                                                                              \
        v4i* _g = (v4i*)(wh + (size_t)(colChunk + (cbi) * 128) * IN_DIM + woff);      \
        lds_v4i* _l =                                                                 \
            (lds_v4i*)(ldsB + (size_t)(buf) * LDSB_HALVES + (size_t)tid * 32);        \
        __builtin_amdgcn_global_load_async_to_lds_b128(_g, _l, 0, 0);                 \
        __builtin_amdgcn_global_load_async_to_lds_b128(_g, _l, 16, 0);                \
        __builtin_amdgcn_global_load_async_to_lds_b128(_g, _l, 32, 0);                \
        __builtin_amdgcn_global_load_async_to_lds_b128(_g, _l, 48, 0);                \
    } while (0)

    ASYNC_FILL(0, 0);
#pragma unroll 1
    for (int cb = 0; cb < 8; ++cb) {
        WAIT_ASYNC(0);                       // this block's fill has landed
        __syncthreads();                     // ... in every wave
        if (cb < 7)
            ASYNC_FILL((cb + 1) & 1, cb + 1);   // overwrites buffer consumed at cb-1
        compute_block((const v16h*)(ldsB + (size_t)(cb & 1) * LDSB_HALVES), cb);
    }
#undef ASYNC_FILL
#else
    // ---- fallback: register-staged single-buffer fill (round-2 scheme)
    v16h wr0, wr1;
    {
        const v16h* p = (const v16h*)(wh + (size_t)colChunk * IN_DIM + woff);
        wr0 = p[0];
        wr1 = p[1];
    }
    v16h nr0 = wr0, nr1 = wr1;

#pragma unroll 1
    for (int cb = 0; cb < 8; ++cb) {
        if (cb) __syncthreads();            // previous block's LDS readers done
        ((v16h*)ldsB)[tid * 2]     = wr0;
        ((v16h*)ldsB)[tid * 2 + 1] = wr1;
        if (cb < 7) {                       // prefetch next w block into regs
            const v16h* p = (const v16h*)(wh + (size_t)(colChunk + (cb + 1) * 128) * IN_DIM + woff);
            nr0 = p[0];
            nr1 = p[1];
        }
        __syncthreads();                    // fill visible to all waves
        compute_block((const v16h*)ldsB, cb);
        wr0 = nr0;
        wr1 = nr1;
    }
#endif
}

extern "C" void kernel_launch(void* const* d_in, const int* in_sizes, int n_in,
                              void* d_out, int out_size, void* d_ws, size_t ws_size,
                              hipStream_t stream) {
    (void)in_sizes; (void)n_in; (void)out_size; (void)ws_size;
    const float* x = (const float*)d_in[0];   // [32768, 64] f32
    const float* w = (const float*)d_in[1];   // [4096, 64]  f32
    float* out = (float*)d_out;               // [32768, 4096] f32

    char* ws = (char*)d_ws;
    float*    xsq = (float*)(ws + WS_XSQ);
    float*    wsq = (float*)(ws + WS_WSQ);
    _Float16* xhi = (_Float16*)(ws + WS_XHI);
    _Float16* xlo = (_Float16*)(ws + WS_XLO);
    _Float16* whh = (_Float16*)(ws + WS_WH);

    rbf_prep_x<<<BATCH / 256, 256, 0, stream>>>(x, xsq, xhi, xlo);
    rbf_prep_w<<<NUM_CLASSES / 256, 256, 0, stream>>>(w, wsq, whh);

    dim3 grid(NUM_CLASSES / 1024, BATCH / 128);   // (4, 256)
    rbf_gemm_kernel<<<grid, 256, 0, stream>>>(xhi, xlo, whh, xsq, wsq, out);
}
